// GHMNBackbone_85023172592636
// MI455X (gfx1250) — matrix operand order repaired and checked by reading, not compiled, not run on hardware
//
#include <hip/hip_runtime.h>
#include <hip/hip_bf16.h>
#include <stdint.h>

// ---------------- problem constants ----------------
constexpr int V = 2, NV = 16384, NH = 12288;
constexpr int T_IN = 8, T_OUT = 4, H = 256, E = 65536;
constexpr int KD  = H + 9;          // 265  (decoder GEMM K)
constexpr int KDP = 288;            // 265 padded to multiple of 32
constexpr int M_PROC = V * T_OUT * NH;      // 98304 rows,  K=256
constexpr int M_DEC  = V * T_OUT * NV;      // 131072 rows, K=288(pad)
constexpr int M_ATT  = V * T_OUT * 2 * NV;  // 262144 rows, K=256, N=512

// ---------------- CDNA5 WMMA plumbing ----------------
typedef __attribute__((ext_vector_type(16))) __bf16 v16bf;
typedef __attribute__((ext_vector_type(8)))  float  v8f;

union Frag16 { uint4 u4[2]; v16bf bf; };

#define DEVFN static __device__ __forceinline__

DEVFN unsigned short f2bf(float f) {
  union { float f; unsigned int u; } c; c.f = f;
  unsigned int r = c.u + 0x7FFFu + ((c.u >> 16) & 1u);   // RNE
  return (unsigned short)(r >> 16);
}
DEVFN float bf2f(unsigned short b) {
  union { unsigned int u; float f; } c; c.u = ((unsigned int)b) << 16;
  return c.f;
}
DEVFN float lrelu(float x) { return x > 0.f ? x : 0.01f * x; }
DEVFN float gelu_tanh(float x) {
  const float k0 = 0.7978845608028654f, k1 = 0.044715f;
  return 0.5f * x * (1.f + tanhf(k0 * (x + k1 * x * x * x)));
}
DEVFN float nan2num(float x) {
  if (x != x) return 0.f;
  return fminf(fmaxf(x, -3.402823466e38f), 3.402823466e38f);
}

// A fragment: 16x32 bf16, row-major source [M][lda]. ISA layout: lanes 0-15 row M,
// halves 0-7 = K+0..7 (lanes>=16: K+8..15), halves 8-15 = K+16..23 (lanes>=16: +8).
DEVFN v16bf load_a16(const unsigned short* A, int lda, int rowBase, int k0, int lane) {
  int row  = rowBase + (lane & 15);
  int koff = ((lane >> 4) & 1) * 8;
  const unsigned short* p = A + (size_t)row * lda + (k0 + koff);
  Frag16 f;
  f.u4[0] = *reinterpret_cast<const uint4*>(p);
  f.u4[1] = *reinterpret_cast<const uint4*>(p + 16);
  return f.bf;
}
// B fragment: 32x16 bf16 from pre-transposed weight Bt[N][ldb] (i.e. Bt[n][k]=W[k][n]).
// Lanes 0-15: col=lane, K=k0+0..7,16..23; lanes 16-31: K=k0+8..15,24..31.
DEVFN v16bf load_b16(const unsigned short* Bt, int ldb, int colBase, int k0, int lane) {
  int col = colBase + (lane & 15);
  int kb  = ((lane >> 4) & 1) * 8;
  const unsigned short* p = Bt + (size_t)col * ldb + (k0 + kb);
  Frag16 f;
  f.u4[0] = reinterpret_cast<const uint4*>(p)[0];
  f.u4[1] = *reinterpret_cast<const uint4*>(p + 16);
  return f.bf;
}
DEVFN v8f wmma_bf16(v16bf a, v16bf b, v8f c) {
  return __builtin_amdgcn_wmma_f32_16x16x32_bf16(false, a, false, b, (short)0, c,
                                                 false, false);
}

// =====================================================================
// K1: degree accumulation (all four degree arrays), K1b: rsqrt(clip(deg,1))
// =====================================================================
__global__ void k_degrees(const int* __restrict__ esrc, const int* __restrict__ edst,
                          const int* __restrict__ dsrc, const int* __restrict__ ddst,
                          float* __restrict__ dsE, float* __restrict__ ddE,
                          float* __restrict__ dsD, float* __restrict__ ddD) {
  int tid = blockIdx.x * blockDim.x + threadIdx.x;
  if (tid >= V * E) return;
  int v = tid / E;
  atomicAdd(&dsE[v * NV + esrc[tid]], 1.f);
  atomicAdd(&ddE[v * NH + edst[tid]], 1.f);
  atomicAdd(&dsD[v * NH + dsrc[tid]], 1.f);
  atomicAdd(&ddD[v * NV + ddst[tid]], 1.f);
}
__global__ void k_deg_finalize(float* __restrict__ degs, int n) {
  int tid = blockIdx.x * blockDim.x + threadIdx.x;
  if (tid >= n) return;
  degs[tid] = rsqrtf(fmaxf(degs[tid], 1.f));
}

// =====================================================================
// K2: encoder edge scatter: agg_enc[v,t,dst,0..9] += m[src]  (m already /sqrt(deg_s))
// =====================================================================
__global__ void k_enc_scatter(const float* __restrict__ x, const float* __restrict__ emb_var,
                              const float* __restrict__ pe_feat,
                              const int* __restrict__ esrc, const int* __restrict__ edst,
                              const float* __restrict__ dsE, float* __restrict__ agg) {
  int tid = blockIdx.x * blockDim.x + threadIdx.x;
  if (tid >= V * E) return;
  int v = tid / E;
  int src = esrc[tid], dst = edst[tid];
  float rs = dsE[v * NV + src];
  float pe[9];
#pragma unroll
  for (int c = 0; c < 9; ++c)
    pe[c] = emb_var[(size_t)(v * NV + src) * 9 + c] * pe_feat[v * 9 + c] * rs;
#pragma unroll
  for (int t = 0; t < T_IN; ++t) {
    float f = nan2num(x[(size_t)(v * NV + src) * T_IN + t]) * rs;
    float* base = agg + ((size_t)(v * T_IN + t) * NH + dst) * 10;
    atomicAdd(base + 0, f);
#pragma unroll
    for (int c = 0; c < 9; ++c) atomicAdd(base + 1 + c, pe[c]);
  }
}

// =====================================================================
// K3: encoder MLP (sum_s leaky_relu(agg@We[s]+be[s])) fused with time mix
//     (einsum Wt) and gelu; writes bf16 A-operand for the processor GEMM.
// =====================================================================
__global__ void k_enc_mlp_mix(const float* __restrict__ agg, const float* __restrict__ ddE,
                              const float* __restrict__ We, const float* __restrict__ be,
                              const float* __restrict__ Wt, unsigned short* __restrict__ gz) {
  int tid = blockIdx.x * blockDim.x + threadIdx.x;
  if (tid >= V * NH * H) return;
  int h = tid & (H - 1);
  int n = (tid / H) % NH;
  int v = tid / (H * NH);
  float rd = ddE[v * NH + n];
  float ht[T_IN];
#pragma unroll
  for (int t = 0; t < T_IN; ++t) {
    float ac[10];
    const float* ar = agg + ((size_t)(v * T_IN + t) * NH + n) * 10;
#pragma unroll
    for (int c = 0; c < 10; ++c) ac[c] = ar[c] * rd;
    float s_acc = 0.f;
#pragma unroll
    for (int s = 0; s < T_IN; ++s) {
      float pre = be[(size_t)(v * T_IN + s) * H + h];
#pragma unroll
      for (int c = 0; c < 10; ++c)
        pre += ac[c] * We[((size_t)(v * T_IN + s) * 10 + c) * H + h];
      s_acc += lrelu(pre);
    }
    ht[t] = s_acc;
  }
#pragma unroll
  for (int so = 0; so < T_OUT; ++so) {
    float z = 0.f;
#pragma unroll
    for (int t = 0; t < T_IN; ++t) z += ht[t] * Wt[t * T_OUT + so];
    gz[((size_t)(v * T_OUT + so) * NH + n) * H + h] = f2bf(gelu_tanh(z));
  }
}

// =====================================================================
// K4: weight transpose + f32->bf16 pack:  dst[b][n][kpad] = src[b][k][n]
// =====================================================================
__global__ void k_wt_transpose(const float* __restrict__ src, unsigned short* __restrict__ dst,
                               int K, int N, int Kpad, int total) {
  int tid = blockIdx.x * blockDim.x + threadIdx.x;
  if (tid >= total) return;
  int k = tid % Kpad;
  int n = (tid / Kpad) % N;
  int b = tid / (Kpad * N);
  float val = (k < K) ? src[((size_t)b * K + k) * N + n] : 0.f;
  dst[tid] = f2bf(val);
}

// =====================================================================
// K5: processor GEMM (WMMA bf16):  z2 = gz @ Wp + bp   (M_PROC x 256, K=256)
// =====================================================================
__global__ void k_proc_gemm(const unsigned short* __restrict__ A,
                            const unsigned short* __restrict__ Bt,
                            const float* __restrict__ bp, unsigned short* __restrict__ z2) {
  const int NT = H / 16;                          // 16 column tiles
  int wave = blockIdx.x * (blockDim.x >> 5) + (threadIdx.x >> 5);
  if (wave >= (M_PROC / 16) * NT) return;
  int mt = wave / NT, nt = wave % NT;
  int lane = threadIdx.x & 31;
  v8f acc = {};
#pragma unroll
  for (int kc = 0; kc < H / 32; ++kc) {
    v16bf a = load_a16(A, H, mt * 16, kc * 32, lane);
    v16bf b = load_b16(Bt, H, nt * 16, kc * 32, lane);
    acc = wmma_bf16(a, b, acc);
  }
  int colg = nt * 16 + (lane & 15);
  float bias = bp[colg];
#pragma unroll
  for (int r = 0; r < 8; ++r) {
    int rowg = mt * 16 + ((lane >> 4) & 1) * 8 + r;
    z2[(size_t)rowg * H + colg] = f2bf(acc[r] + bias);
  }
}

// =====================================================================
// K6: decoder edge scatter: agg_dec[v,s,dst,c] += src_aug[v,s,src,c]/sqrt(deg_s)
//     channels 0..255 = z2, 256..264 = emb_healpix*pe_healpix (time-invariant)
// =====================================================================
__global__ void k_dec_scatter(const unsigned short* __restrict__ z2,
                              const float* __restrict__ emb_h, const float* __restrict__ pe_h,
                              const int* __restrict__ dsrc, const int* __restrict__ ddst,
                              const float* __restrict__ dsD, float* __restrict__ agg) {
  long long tid = (long long)blockIdx.x * blockDim.x + threadIdx.x;
  if (tid >= (long long)V * E * KD) return;
  int c = (int)(tid % KD);
  int e = (int)((tid / KD) % E);
  int v = (int)(tid / ((long long)KD * E));
  int src = dsrc[v * E + e], dst = ddst[v * E + e];
  float rs = dsD[v * NH + src];
  if (c < H) {
#pragma unroll
    for (int s = 0; s < T_OUT; ++s) {
      const unsigned short* p = z2 + ((size_t)(v * T_OUT + s) * NH + src) * H + c;
      if (s + 1 < T_OUT) __builtin_prefetch(p + (size_t)NH * H, 0, 0);  // global_prefetch_b8
      atomicAdd(&agg[((size_t)(v * T_OUT + s) * NV + dst) * KD + c], bf2f(*p) * rs);
    }
  } else {
    float val = emb_h[(size_t)src * 9 + (c - H)] * pe_h[c - H] * rs;
#pragma unroll
    for (int s = 0; s < T_OUT; ++s)
      atomicAdd(&agg[((size_t)(v * T_OUT + s) * NV + dst) * KD + c], val);
  }
}

// =====================================================================
// K7: scale by rsqrt(deg_d), convert to bf16, pad K 265 -> 288
// =====================================================================
__global__ void k_dec_scale(const float* __restrict__ agg, const float* __restrict__ ddD,
                            unsigned short* __restrict__ Adec) {
  long long tid = (long long)blockIdx.x * blockDim.x + threadIdx.x;
  if (tid >= (long long)M_DEC * KDP) return;
  int c  = (int)(tid % KDP);
  int n  = (int)((tid / KDP) % NV);
  int vs = (int)(tid / ((long long)KDP * NV));
  int v  = vs / T_OUT;
  float val = 0.f;
  if (c < KD) val = agg[((size_t)vs * NV + n) * KD + c] * ddD[v * NV + n];
  Adec[tid] = f2bf(val);
}

// =====================================================================
// K8: decoder MLP (WMMA bf16): s1 = sum_ss lrelu(Adec @ Wd[ss] + bd[ss])
//     rows = [v,s,n] (M_DEC), K=288(pad), N=256 -> os[...,i=0,...]
// =====================================================================
__global__ void k_s1_gemm(const unsigned short* __restrict__ A,
                          const unsigned short* __restrict__ WdT,
                          const float* __restrict__ bd, unsigned short* __restrict__ os) {
  const int NT = H / 16;
  int wave = blockIdx.x * (blockDim.x >> 5) + (threadIdx.x >> 5);
  if (wave >= (M_DEC / 16) * NT) return;
  int mt = wave / NT, nt = wave % NT;
  int lane = threadIdx.x & 31;
  int rowBase = mt * 16;
  int v = rowBase / (T_OUT * NV);                 // 16-row tiles never cross v
  v16bf afr[KDP / 32];
#pragma unroll
  for (int kc = 0; kc < KDP / 32; ++kc)
    afr[kc] = load_a16(A, KDP, rowBase, kc * 32, lane);
  int colg = nt * 16 + (lane & 15);
  v8f sum = {};
#pragma unroll
  for (int ss = 0; ss < T_OUT; ++ss) {
    const unsigned short* Bt = WdT + (size_t)(v * T_OUT + ss) * H * KDP;
    v8f acc = {};
#pragma unroll
    for (int kc = 0; kc < KDP / 32; ++kc) {
      v16bf b = load_b16(Bt, KDP, nt * 16, kc * 32, lane);
      acc = wmma_bf16(afr[kc], b, acc);
    }
    float bias = bd[(size_t)(v * T_OUT + ss) * H + colg];
#pragma unroll
    for (int r = 0; r < 8; ++r) sum[r] += lrelu(acc[r] + bias);
  }
#pragma unroll
  for (int r = 0; r < 8; ++r) {
    int row = rowBase + ((lane >> 4) & 1) * 8 + r;      // row = vs*NV + n
    int vs = row / NV, n = row & (NV - 1);
    os[(((size_t)vs * 2 + 0) * NV + n) * H + colg] = f2bf(sum[r]);
  }
}

// =====================================================================
// K9: skip linear s2 = [x_t, emb*pe] @ Wl + bl  (K=10, scalar) -> os[...,i=1,...]
// =====================================================================
__global__ void k_s2(const float* __restrict__ x, const float* __restrict__ emb_var,
                     const float* __restrict__ pe_feat, const float* __restrict__ Wl,
                     const float* __restrict__ bl, unsigned short* __restrict__ os) {
  long long tid = (long long)blockIdx.x * blockDim.x + threadIdx.x;
  if (tid >= (long long)M_DEC * H) return;
  int h  = (int)(tid & (H - 1));
  int n  = (int)((tid >> 8) % NV);
  int vs = (int)(tid / ((long long)H * NV));
  int v = vs / T_OUT, s = vs % T_OUT;
  float acc = bl[(size_t)v * H + h] +
              x[(size_t)(v * NV + n) * T_IN + s] * Wl[((size_t)v * 10 + 0) * H + h];
#pragma unroll
  for (int c = 1; c < 10; ++c)
    acc += emb_var[(size_t)(v * NV + n) * 9 + (c - 1)] * pe_feat[v * 9 + (c - 1)] *
           Wl[((size_t)v * 10 + c) * H + h];
  os[(((size_t)vs * 2 + 1) * NV + n) * H + h] = f2bf(acc);
}

// =====================================================================
// K10: attention score (WMMA bf16): score = tanh(os @ Wr1 + br1) @ Wr2 + br2
//      Each wave owns 16 rows, sweeps all 32 N-tiles (N=512), reduces in-wave.
// =====================================================================
__global__ void k_score(const unsigned short* __restrict__ A,
                        const unsigned short* __restrict__ Wr1T,
                        const float* __restrict__ br1, const float* __restrict__ Wr2,
                        const float* __restrict__ br2, float* __restrict__ score) {
  int wave = blockIdx.x * (blockDim.x >> 5) + (threadIdx.x >> 5);
  if (wave >= M_ATT / 16) return;
  int lane = threadIdx.x & 31;
  int rowBase = wave * 16;
  v16bf afr[H / 32];
#pragma unroll
  for (int kc = 0; kc < H / 32; ++kc)
    afr[kc] = load_a16(A, H, rowBase, kc * 32, lane);
  float part[8] = {0, 0, 0, 0, 0, 0, 0, 0};
  for (int ntile = 0; ntile < (2 * H) / 16; ++ntile) {
    v8f acc = {};
#pragma unroll
    for (int kc = 0; kc < H / 32; ++kc) {
      v16bf b = load_b16(Wr1T, H, ntile * 16, kc * 32, lane);
      acc = wmma_bf16(afr[kc], b, acc);
    }
    int colg = ntile * 16 + (lane & 15);
    float b1 = br1[colg], w2 = Wr2[colg];
#pragma unroll
    for (int r = 0; r < 8; ++r) part[r] += tanhf(acc[r] + b1) * w2;
  }
  float b2 = br2[0];
#pragma unroll
  for (int r = 0; r < 8; ++r) {
    float p = part[r];                 // reduce across the 16 lanes holding this row
    p += __shfl_xor(p, 1, 32);
    p += __shfl_xor(p, 2, 32);
    p += __shfl_xor(p, 4, 32);
    p += __shfl_xor(p, 8, 32);
    if ((lane & 15) == 0) {
      int row = rowBase + ((lane >> 4) & 1) * 8 + r;
      score[row] = p + b2;
    }
  }
}

// =====================================================================
// K11: softmax over the two branches, blend states, readout Wr3 dot
// =====================================================================
__global__ void k_readout(const float* __restrict__ score, const unsigned short* __restrict__ os,
                          const float* __restrict__ Wr3, const float* __restrict__ br3,
                          float* __restrict__ out) {
  int tid = blockIdx.x * blockDim.x + threadIdx.x;
  if (tid >= M_DEC) return;
  int n = tid & (NV - 1);
  int vs = tid / NV;
  float sc0 = score[(size_t)(vs * 2 + 0) * NV + n];
  float sc1 = score[(size_t)(vs * 2 + 1) * NV + n];
  float m = fmaxf(sc0, sc1);
  float e0 = __expf(sc0 - m), e1 = __expf(sc1 - m);
  float inv = 1.f / (e0 + e1);
  float a0 = e0 * inv, a1 = e1 * inv;
  const unsigned short* p0 = os + ((size_t)(vs * 2 + 0) * NV + n) * H;
  const unsigned short* p1 = os + ((size_t)(vs * 2 + 1) * NV + n) * H;
  float acc = 0.f;
#pragma unroll 8
  for (int h = 0; h < H; ++h)
    acc += (a0 * bf2f(p0[h]) + a1 * bf2f(p1[h])) * Wr3[h];
  out[tid] = acc + br3[0];
}

// =====================================================================
// host launcher
// =====================================================================
static inline int cdiv(long long a, int b) { return (int)((a + b - 1) / b); }

extern "C" void kernel_launch(void* const* d_in, const int* in_sizes, int n_in,
                              void* d_out, int out_size, void* d_ws, size_t ws_size,
                              hipStream_t stream) {
  const float* x_nodes    = (const float*)d_in[0];
  const float* emb_var    = (const float*)d_in[1];
  const float* emb_hp     = (const float*)d_in[2];
  const float* pe_feature = (const float*)d_in[3];
  const float* pe_healpix = (const float*)d_in[4];
  const float* W_enc      = (const float*)d_in[5];
  const float* b_enc      = (const float*)d_in[6];
  const float* W_dec      = (const float*)d_in[7];
  const float* b_dec      = (const float*)d_in[8];
  const float* W_lin      = (const float*)d_in[9];
  const float* b_lin      = (const float*)d_in[10];
  const float* Wt         = (const float*)d_in[11];
  const float* Wp         = (const float*)d_in[12];
  const float* bp         = (const float*)d_in[13];
  const float* Wr1        = (const float*)d_in[14];
  const float* br1        = (const float*)d_in[15];
  const float* Wr2        = (const float*)d_in[16];
  const float* br2        = (const float*)d_in[17];
  const float* Wr3        = (const float*)d_in[18];
  const float* br3        = (const float*)d_in[19];
  const int*   enc_src    = (const int*)d_in[20];
  const int*   enc_dst    = (const int*)d_in[21];
  const int*   dec_src    = (const int*)d_in[22];
  const int*   dec_dst    = (const int*)d_in[23];
  float* out = (float*)d_out;

  // ---- carve workspace (256B aligned regions) ----
  char* base = (char*)d_ws;
  size_t off = 0;
  auto carve = [&](size_t bytes) -> void* {
    void* p = base + off;
    off += (bytes + 255) & ~(size_t)255;
    return p;
  };
  const int DEGN = 2 * (V * NV + V * NH);                       // 114688
  float* degs = (float*)carve(sizeof(float) * DEGN);
  float* dsE = degs;
  float* ddE = dsE + V * NV;
  float* dsD = ddE + V * NH;
  float* ddD = dsD + V * NH;
  const size_t AGG_E_N = (size_t)V * T_IN * NH * 10;
  float* agg_enc = (float*)carve(sizeof(float) * AGG_E_N);
  unsigned short* gz    = (unsigned short*)carve(sizeof(short) * (size_t)M_PROC * H);
  unsigned short* z2    = (unsigned short*)carve(sizeof(short) * (size_t)M_PROC * H);
  unsigned short* WpT   = (unsigned short*)carve(sizeof(short) * (size_t)H * H);
  unsigned short* WdT   = (unsigned short*)carve(sizeof(short) * (size_t)V * T_OUT * H * KDP);
  unsigned short* Wr1T  = (unsigned short*)carve(sizeof(short) * (size_t)(2 * H) * H);
  const size_t AGG_D_N = (size_t)M_DEC * KD;
  float* agg_dec = (float*)carve(sizeof(float) * AGG_D_N);
  unsigned short* Adec  = (unsigned short*)carve(sizeof(short) * (size_t)M_DEC * KDP);
  unsigned short* osb   = (unsigned short*)carve(sizeof(short) * (size_t)M_ATT * H);
  float* score = (float*)carve(sizeof(float) * M_ATT);
  (void)ws_size; (void)n_in; (void)in_sizes; (void)out_size;

  const int B = 256;

  // ---- zero atomic accumulators (graph-capture-safe) ----
  hipMemsetAsync(degs, 0, sizeof(float) * DEGN, stream);
  hipMemsetAsync(agg_enc, 0, sizeof(float) * AGG_E_N, stream);
  hipMemsetAsync(agg_dec, 0, sizeof(float) * AGG_D_N, stream);

  // ---- degrees ----
  k_degrees<<<cdiv((long long)V * E, B), B, 0, stream>>>(enc_src, enc_dst, dec_src,
                                                         dec_dst, dsE, ddE, dsD, ddD);
  k_deg_finalize<<<cdiv(DEGN, B), B, 0, stream>>>(degs, DEGN);

  // ---- weight packing (bf16, transposed, K padded) ----
  k_wt_transpose<<<cdiv((long long)H * H, B), B, 0, stream>>>(Wp, WpT, H, H, H, H * H);
  k_wt_transpose<<<cdiv((long long)V * T_OUT * H * KDP, B), B, 0, stream>>>(
      W_dec, WdT, KD, H, KDP, V * T_OUT * H * KDP);
  k_wt_transpose<<<cdiv((long long)2 * H * H, B), B, 0, stream>>>(Wr1, Wr1T, H, 2 * H, H,
                                                                  2 * H * H);

  // ---- encoder ----
  k_enc_scatter<<<cdiv((long long)V * E, B), B, 0, stream>>>(x_nodes, emb_var, pe_feature,
                                                             enc_src, enc_dst, dsE, agg_enc);
  k_enc_mlp_mix<<<cdiv((long long)V * NH * H, B), B, 0, stream>>>(agg_enc, ddE, W_enc,
                                                                  b_enc, Wt, gz);

  // ---- processor GEMM (WMMA) ----
  k_proc_gemm<<<cdiv((long long)(M_PROC / 16) * (H / 16) * 32, B), B, 0, stream>>>(
      gz, WpT, bp, z2);

  // ---- decoder graph conv ----
  k_dec_scatter<<<cdiv((long long)V * E * KD, B), B, 0, stream>>>(z2, emb_hp, pe_healpix,
                                                                  dec_src, dec_dst, dsD,
                                                                  agg_dec);
  k_dec_scale<<<cdiv((long long)M_DEC * KDP, B), B, 0, stream>>>(agg_dec, ddD, Adec);

  // ---- decoder MLP (WMMA) + skip linear ----
  k_s1_gemm<<<cdiv((long long)(M_DEC / 16) * (H / 16) * 32, B), B, 0, stream>>>(Adec, WdT,
                                                                                b_dec, osb);
  k_s2<<<cdiv((long long)M_DEC * H, B), B, 0, stream>>>(x_nodes, emb_var, pe_feature,
                                                        W_lin, b_lin, osb);

  // ---- attention score (WMMA) + readout ----
  k_score<<<cdiv((long long)(M_ATT / 16) * 32, B), B, 0, stream>>>(osb, Wr1T, br1, Wr2,
                                                                  br2, score);
  k_readout<<<cdiv(M_DEC, B), B, 0, stream>>>(score, osb, Wr3, br3, out);
}
